// GriffinTemporalBlock_17394617549236
// MI455X (gfx1250) — matrix operand order repaired and checked by reading, not compiled, not run on hardware
//
#include <hip/hip_runtime.h>

typedef __attribute__((ext_vector_type(16))) __bf16 v16bf;
typedef __attribute__((ext_vector_type(8)))  float  v8f;
typedef __attribute__((ext_vector_type(4)))  int    v4i_;

#define BB 16
#define TT 1024
#define DD 1024
#define HH 4096
#define NR (BB * TT)   // 16384 rows

// ---------- bf16 helpers (round-to-nearest-even) ----------
__device__ __forceinline__ unsigned short f2bf(float f) {
  unsigned int u = __float_as_uint(f);
  u += 0x7fffu + ((u >> 16) & 1u);
  return (unsigned short)(u >> 16);
}
__device__ __forceinline__ float bf2f(unsigned short h) {
  return __uint_as_float(((unsigned int)h) << 16);
}

// ---------- CDNA5 async global->LDS staging (guarded, with sync fallback) ----
#if __has_builtin(__builtin_amdgcn_global_load_async_to_lds_b128)
#define ASYNC_LDS 1
typedef __attribute__((address_space(1))) v4i_ GQ;   // global int4
typedef __attribute__((address_space(3))) v4i_ LQ;   // LDS int4
#else
#define ASYNC_LDS 0
#endif

__device__ __forceinline__ void stage_wait() {
#if ASYNC_LDS
#if __has_builtin(__builtin_amdgcn_s_wait_asynccnt)
  __builtin_amdgcn_s_wait_asynccnt(0);
#else
  asm volatile("s_wait_asynccnt 0" ::: "memory");
#endif
#endif
}

// copy 64 contiguous bytes: global(src) -> LDS(dst), per calling thread
__device__ __forceinline__ void stage_row(const unsigned short* __restrict__ src,
                                          unsigned short* dst) {
#if ASYNC_LDS
  GQ* g = (GQ*)(const void*)src;
  LQ* l = (LQ*)dst;
  __builtin_amdgcn_global_load_async_to_lds_b128(g, l, 0, 0);
  __builtin_amdgcn_global_load_async_to_lds_b128(g, l, 16, 0);
  __builtin_amdgcn_global_load_async_to_lds_b128(g, l, 32, 0);
  __builtin_amdgcn_global_load_async_to_lds_b128(g, l, 48, 0);
#else
  const uint4* s = (const uint4*)src;
  uint4* d = (uint4*)dst;
  d[0] = s[0]; d[1] = s[1]; d[2] = s[2]; d[3] = s[3];
#endif
}

// =========================================================================
// NT GEMM: C[M,N] = A[M,K] (bf16 rowmajor) * W[N,K]^T (bf16 rowmajor)
// 256 threads = 8 waves, block tile 128x128, K staged 32 deep, double-buffered
// LDS filled by async global->LDS DMA. Wave (wm,wn) computes 64x32 via 4x2
// v_wmma_f32_16x16x32_bf16.
// EPI: 0 = f32, 1 = bias+sigmoid f32, 2 = gelu bf16, 3 = bf16, 4 = f32+resid
// =========================================================================
template <int EPI>
__global__ __launch_bounds__(256) void gemm_bf16_nt(
    const unsigned short* __restrict__ A, const unsigned short* __restrict__ W,
    int M, int N, int K,
    float* __restrict__ Cf, unsigned short* __restrict__ Cb,
    const float* __restrict__ bias, const float* __restrict__ resid) {
  __shared__ unsigned short lA[2][128][40];  // 32 K-halves + 8 pad
  __shared__ unsigned short lB[2][128][40];

  const int tid  = threadIdx.x;
  const int lane = tid & 31;
  const int w    = tid >> 5;
  const int wm   = w & 1;
  const int wn   = w >> 1;
  const int l15  = lane & 15;
  const int hi   = lane >> 4;

  const int m0 = blockIdx.y * 128;
  const int n0 = blockIdx.x * 128;

  const v8f vzero = {0.f, 0.f, 0.f, 0.f, 0.f, 0.f, 0.f, 0.f};
  v8f acc[4][2];
#pragma unroll
  for (int i = 0; i < 4; ++i)
#pragma unroll
    for (int j = 0; j < 2; ++j) acc[i][j] = vzero;

  // cooperative staging: threads 0..127 stage A rows, 128..255 stage W rows
  const int  r   = tid & 127;
  const bool isB = tid >= 128;
  const unsigned short* srcRow =
      isB ? (W + (size_t)(n0 + r) * K) : (A + (size_t)(m0 + r) * K);
  unsigned short* dstBuf[2] = {isB ? &lB[0][r][0] : &lA[0][r][0],
                               isB ? &lB[1][r][0] : &lA[1][r][0]};

  union Frag { v16bf v; uint4 q[2]; };

  const int nk = K / 32;
  // prologue: fill buffer 0
  stage_row(srcRow, dstBuf[0]);
  stage_wait();
  __syncthreads();

  for (int ik = 0; ik < nk; ++ik) {
    const int cur = ik & 1;
    if (ik + 1 < nk) stage_row(srcRow + (ik + 1) * 32, dstBuf[cur ^ 1]);

    // B fragments: lane holds column n = l15, K = hi*16 .. hi*16+15 contiguous
    Frag bf[2];
#pragma unroll
    for (int nt = 0; nt < 2; ++nt) {
      const uint4* p = (const uint4*)&lB[cur][wn * 32 + nt * 16 + l15][hi * 16];
      bf[nt].q[0] = p[0];
      bf[nt].q[1] = p[1];
    }
#pragma unroll
    for (int mt = 0; mt < 4; ++mt) {
      // A fragment: lane row = l15; K = hi*16 .. hi*16+15 contiguous
      Frag af;
      af.q[0] = *(const uint4*)&lA[cur][wm * 64 + mt * 16 + l15][hi * 16];
      af.q[1] = *(const uint4*)&lA[cur][wm * 64 + mt * 16 + l15][hi * 16 + 8];
#pragma unroll
      for (int nt = 0; nt < 2; ++nt) {
        acc[mt][nt] = __builtin_amdgcn_wmma_f32_16x16x32_bf16(
            false, af.v, false, bf[nt].v, (short)0, acc[mt][nt], false, false);
      }
    }

    if (ik + 1 < nk) {
      stage_wait();
      __syncthreads();
    }
  }

  // epilogue: C element (vgpr v): m = v + hi*8, n = l15
#pragma unroll
  for (int mt = 0; mt < 4; ++mt) {
#pragma unroll
    for (int nt = 0; nt < 2; ++nt) {
#pragma unroll
      for (int v = 0; v < 8; ++v) {
        int m = m0 + wm * 64 + mt * 16 + hi * 8 + v;
        int n = n0 + wn * 32 + nt * 16 + l15;
        size_t o = (size_t)m * N + n;
        float x = acc[mt][nt][v];
        if (EPI == 0) {
          Cf[o] = x;
        } else if (EPI == 1) {
          x += bias[n];
          Cf[o] = 1.f / (1.f + expf(-x));
        } else if (EPI == 2) {
          Cb[o] = f2bf(0.5f * x * (1.f + erff(x * 0.70710678118654752f)));
        } else if (EPI == 3) {
          Cb[o] = f2bf(x);
        } else {
          Cf[o] = x + resid[o];
        }
      }
    }
  }
}

// ---------- elementwise kernels ----------
__global__ void cvt_bf16_kernel(const float* __restrict__ s,
                                unsigned short* __restrict__ d, int n) {
  int i = blockIdx.x * 256 + threadIdx.x;
  if (i < n) d[i] = f2bf(s[i]);
}

__global__ void rmsnorm_kernel(const float* __restrict__ x,
                               const float* __restrict__ w,
                               unsigned short* __restrict__ out) {
  __shared__ float red[256];
  const int row = blockIdx.x;
  const float* xr = x + (size_t)row * DD;
  float s = 0.f;
  for (int i = threadIdx.x; i < DD; i += 256) {
    float v = xr[i];
    s += v * v;
  }
  red[threadIdx.x] = s;
  __syncthreads();
  for (int off = 128; off > 0; off >>= 1) {
    if (threadIdx.x < off) red[threadIdx.x] += red[threadIdx.x + off];
    __syncthreads();
  }
  float scale = rsqrtf(red[0] * (1.0f / DD) + 1.1920929e-07f);
  for (int i = threadIdx.x; i < DD; i += 256)
    out[(size_t)row * DD + i] = f2bf(xr[i] * scale * w[i]);
}

__global__ void conv_kernel(const float* __restrict__ b1,
                            const float* __restrict__ conv_buf,
                            const float* __restrict__ conv_w,
                            const float* __restrict__ conv_b,
                            float* __restrict__ outf,
                            unsigned short* __restrict__ outb) {
  size_t i = (size_t)blockIdx.x * 256 + threadIdx.x;  // over NR*DD
  int d = (int)(i % DD);
  size_t bt = i / DD;
  int t = (int)(bt % TT);
  int b = (int)(bt / TT);
  float acc = conv_b[d];
#pragma unroll
  for (int k = 0; k < 4; ++k) {
    int p = t + k;  // padded index
    float v = (p < 3) ? conv_buf[((size_t)b * 3 + p) * DD + d]
                      : b1[((size_t)b * TT + (p - 3)) * DD + d];
    acc += v * conv_w[d * 4 + k];
  }
  outf[i] = acc;
  outb[i] = f2bf(acc);
}

__global__ void newbuf_kernel(const float* __restrict__ b1,
                              float* __restrict__ out) {
  int i = blockIdx.x * 256 + threadIdx.x;  // over BB*3*DD
  int d = i % DD;
  int j = (i / DD) % 3;
  int b = i / (3 * DD);
  out[i] = b1[((size_t)b * TT + (TT - 3 + j)) * DD + d];
}

__global__ void gate_kernel(float* __restrict__ r, float* __restrict__ ii,
                            const float* __restrict__ conv,
                            const float* __restrict__ log_lambda) {
  size_t i = (size_t)blockIdx.x * 256 + threadIdx.x;  // over NR*DD
  int d = (int)(i % DD);
  float lam = log_lambda[d];
  float ls = -log1pf(expf(-lam));  // log_sigmoid(lambda)
  float a = expf(8.0f * r[i] * ls);
  float gate = sqrtf(fmaxf((1.f - a) * (1.f + a), 1e-6f));
  float u = gate * ii[i] * conv[i];
  r[i] = a;   // a  (in place over r)
  ii[i] = u;  // u  (in place over i)
}

__global__ void scan_kernel(const float* __restrict__ a,
                            const float* __restrict__ u,
                            const float* __restrict__ h0,
                            const unsigned short* __restrict__ b2,
                            unsigned short* __restrict__ prod,
                            float* __restrict__ newh) {
  int idx = blockIdx.x * 256 + threadIdx.x;  // b*DD + d
  int b = idx / DD;
  int d = idx % DD;
  float h = h0[idx];
  size_t base = (size_t)b * TT * DD + d;
  for (int t = 0; t < TT; ++t) {
    size_t j = base + (size_t)t * DD;
    h = a[j] * h + u[j];
    prod[j] = f2bf(h * bf2f(b2[j]));  // b1_out * b2, bf16 for Wout GEMM
  }
  newh[idx] = h;
}

__global__ void bmul_kernel(unsigned short* __restrict__ g,
                            const unsigned short* __restrict__ u) {
  size_t i = (size_t)blockIdx.x * 256 + threadIdx.x;  // over NR*HH
  g[i] = f2bf(bf2f(g[i]) * bf2f(u[i]));
}

// ---------- host launcher ----------
static void run_gemm(int epi, const unsigned short* A, const unsigned short* W,
                     int M, int N, int K, float* Cf, unsigned short* Cb,
                     const float* bias, const float* resid, hipStream_t st) {
  dim3 g(N / 128, M / 128), blk(256);
  switch (epi) {
    case 0: gemm_bf16_nt<0><<<g, blk, 0, st>>>(A, W, M, N, K, Cf, Cb, bias, resid); break;
    case 1: gemm_bf16_nt<1><<<g, blk, 0, st>>>(A, W, M, N, K, Cf, Cb, bias, resid); break;
    case 2: gemm_bf16_nt<2><<<g, blk, 0, st>>>(A, W, M, N, K, Cf, Cb, bias, resid); break;
    case 3: gemm_bf16_nt<3><<<g, blk, 0, st>>>(A, W, M, N, K, Cf, Cb, bias, resid); break;
    default: gemm_bf16_nt<4><<<g, blk, 0, st>>>(A, W, M, N, K, Cf, Cb, bias, resid); break;
  }
}

extern "C" void kernel_launch(void* const* d_in, const int* in_sizes, int n_in,
                              void* d_out, int out_size, void* d_ws, size_t ws_size,
                              hipStream_t stream) {
  const float* x_seq      = (const float*)d_in[0];
  const float* h0         = (const float*)d_in[1];
  const float* conv_buf   = (const float*)d_in[2];
  const float* norm1_w    = (const float*)d_in[3];
  const float* W1         = (const float*)d_in[4];
  const float* conv_w     = (const float*)d_in[5];
  const float* conv_b     = (const float*)d_in[6];
  const float* Wa         = (const float*)d_in[7];
  const float* ba         = (const float*)d_in[8];
  const float* Wx         = (const float*)d_in[9];
  const float* bx         = (const float*)d_in[10];
  const float* log_lambda = (const float*)d_in[11];
  const float* W2         = (const float*)d_in[12];
  const float* Wout       = (const float*)d_in[13];
  const float* norm2_w    = (const float*)d_in[14];
  const float* Wg         = (const float*)d_in[15];
  const float* Wu         = (const float*)d_in[16];
  const float* Wo         = (const float*)d_in[17];

  const size_t MB = 1ull << 20;
  char* ws = (char*)d_ws;
  // live-range-aliased workspace layout (peak ~514 MB)
  unsigned short* nrmb  = (unsigned short*)(ws + 0);         // 32MB, dead after b2 GEMM
  float*          b1f   = (float*)(ws + 32 * MB);            // 64MB, dead after conv
  float*          convf = (float*)(ws + 96 * MB);            // 64MB, dead after gate
  unsigned short* convb = (unsigned short*)(ws + 160 * MB);  // 32MB, dead after Wx GEMM
  float*          rf    = (float*)(ws + 192 * MB);           // 64MB (r then a), dead after scan
  float*          uf    = (float*)(ws + 256 * MB);           // 64MB (i then u), dead after scan
  unsigned short* b2b   = (unsigned short*)(ws + 320 * MB);  // 32MB, dead after scan
  unsigned short* prodb = (unsigned short*)(ws + 352 * MB);  // 32MB, dead after Wout GEMM
  float*          x1f   = (float*)(ws + 384 * MB);           // 64MB, live to end
  unsigned short* n2b   = (unsigned short*)(ws + 448 * MB);  // 32MB
  unsigned short* Gb    = (unsigned short*)(ws + 0);         // 128MB, reuses [0,160MB)
  unsigned short* Ub    = (unsigned short*)(ws + 160 * MB);  // 128MB, reuses [160,288MB)
  unsigned short* wb    = (unsigned short*)(ws + 480 * MB);  // 34MB bf16 weights
  unsigned short* W1b   = wb;
  unsigned short* Wab   = wb + 1 * 1048576;
  unsigned short* Wxb   = wb + 2 * 1048576;
  unsigned short* W2b   = wb + 3 * 1048576;
  unsigned short* Woutb = wb + 4 * 1048576;
  unsigned short* Wgb   = wb + 5 * 1048576;
  unsigned short* Wub   = Wgb + 4 * 1048576;
  unsigned short* Wob   = Wub + 4 * 1048576;

  float* out   = (float*)d_out;
  float* newh  = out + (size_t)NR * DD;   // new_h
  float* newcb = newh + (size_t)BB * DD;  // new_conv_buf

  // 0) weights -> bf16
  cvt_bf16_kernel<<<4096, 256, 0, stream>>>(W1, W1b, DD * DD);
  cvt_bf16_kernel<<<4096, 256, 0, stream>>>(Wa, Wab, DD * DD);
  cvt_bf16_kernel<<<4096, 256, 0, stream>>>(Wx, Wxb, DD * DD);
  cvt_bf16_kernel<<<4096, 256, 0, stream>>>(W2, W2b, DD * DD);
  cvt_bf16_kernel<<<4096, 256, 0, stream>>>(Wout, Woutb, DD * DD);
  cvt_bf16_kernel<<<16384, 256, 0, stream>>>(Wg, Wgb, HH * DD);
  cvt_bf16_kernel<<<16384, 256, 0, stream>>>(Wu, Wub, HH * DD);
  cvt_bf16_kernel<<<16384, 256, 0, stream>>>(Wo, Wob, DD * HH);

  // 1) normed = rmsnorm(x_seq) -> bf16
  rmsnorm_kernel<<<NR, 256, 0, stream>>>(x_seq, norm1_w, nrmb);
  // 2) b1 = normed @ W1^T (f32 for conv)
  run_gemm(0, nrmb, W1b, NR, DD, DD, b1f, nullptr, nullptr, nullptr, stream);
  // 3) causal depthwise conv (K=4) + new_conv_buf
  conv_kernel<<<(NR * DD) / 256, 256, 0, stream>>>(b1f, conv_buf, conv_w, conv_b, convf, convb);
  newbuf_kernel<<<(BB * 3 * DD) / 256, 256, 0, stream>>>(b1f, newcb);
  // 4) r = sigmoid(conv @ Wa^T + ba); i = sigmoid(conv @ Wx^T + bx)
  run_gemm(1, convb, Wab, NR, DD, DD, rf, nullptr, ba, nullptr, stream);
  run_gemm(1, convb, Wxb, NR, DD, DD, uf, nullptr, bx, nullptr, stream);
  // 5) a, u (in place over r, i)
  gate_kernel<<<(NR * DD) / 256, 256, 0, stream>>>(rf, uf, convf, log_lambda);
  // 6) b2 = gelu(normed @ W2^T) -> bf16
  run_gemm(2, nrmb, W2b, NR, DD, DD, nullptr, b2b, nullptr, nullptr, stream);
  // 7) sequential scan over T, fused with b1_out*b2 product; emits new_h
  scan_kernel<<<(BB * DD) / 256, 256, 0, stream>>>(rf, uf, h0, b2b, prodb, newh);
  // 8) x1 = x_seq + prod @ Wout^T
  run_gemm(4, prodb, Woutb, NR, DD, DD, x1f, nullptr, nullptr, x_seq, stream);
  // 9) n2 = rmsnorm(x1) -> bf16
  rmsnorm_kernel<<<NR, 256, 0, stream>>>(x1f, norm2_w, n2b);
  // 10) G = gelu(n2 @ Wg^T), U = n2 @ Wu^T, gu = G*U (in place)
  run_gemm(2, n2b, Wgb, NR, HH, DD, nullptr, Gb, nullptr, nullptr, stream);
  run_gemm(3, n2b, Wub, NR, HH, DD, nullptr, Ub, nullptr, nullptr, stream);
  bmul_kernel<<<(NR * HH) / 256, 256, 0, stream>>>(Gb, Ub);
  // 11) x2 = x1 + gu @ Wo^T  -> d_out
  run_gemm(4, Gb, Wob, NR, DD, HH, out, nullptr, nullptr, x1f, stream);
}
